// BiMambaBlock_79714593014156
// MI455X (gfx1250) — compile-verified, matrix operand order checked by reading, not run on hardware
//
#include <hip/hip_runtime.h>
#include <cstdint>

#define D_MODEL 1024
#define D_INNER 2048
#define D_STATE 16
#define D_CONV  4
#define DT_RANK 64
#define BATCH   2
#define SEQ     2048
#define MTOK    (BATCH*SEQ)        // 4096 tokens

typedef _Float16 h16;
typedef __attribute__((ext_vector_type(16))) _Float16 v16h;
typedef __attribute__((ext_vector_type(8)))  _Float16 v8h;
typedef __attribute__((ext_vector_type(8)))  float    v8f;

#define WMMA_F16(a, b, c) __builtin_amdgcn_wmma_f32_16x16x32_f16( \
    false, (a), false, (b), (short)0, (c), false, false)

__device__ __forceinline__ float silu_f(float v) { return v / (1.f + __expf(-v)); }

// Async memory->LDS copy of 16 bytes per lane (CDNA5 ASYNCcnt path).
__device__ __forceinline__ void async_b128(uint32_t lds_addr, const void* gptr)
{
    uint64_t ga = (uint64_t)(uintptr_t)gptr;
    asm volatile("global_load_async_to_lds_b128 %0, %1, off"
                 :: "v"(lds_addr), "v"(ga) : "memory");
}
// Retire everything (end of pipeline).
__device__ __forceinline__ void wait_async0()
{
    asm volatile("s_wait_asynccnt 0x0" ::: "memory");
}
// Retire the oldest tile (5 async ops/tile/wave); next tile stays in flight.
__device__ __forceinline__ void wait_async5()
{
    asm volatile("s_wait_asynccnt 0x5" ::: "memory");
}

// ---------------------------------------------------------------------------
// LayerNorm: one block per token row, writes f16 activations for WMMA GEMMs
// ---------------------------------------------------------------------------
__global__ __launch_bounds__(256)
void k_layernorm(const float* __restrict__ x, const float* __restrict__ g,
                 const float* __restrict__ b, h16* __restrict__ hn)
{
    __shared__ float rs[256], rq[256];
    const int m = blockIdx.x, tid = threadIdx.x;
    const float* xr = x + (size_t)m * D_MODEL;
    float s = 0.f, q = 0.f;
    for (int c = tid; c < D_MODEL; c += 256) { float v = xr[c]; s += v; q += v * v; }
    rs[tid] = s; rq[tid] = q; __syncthreads();
    for (int st = 128; st > 0; st >>= 1) {
        if (tid < st) { rs[tid] += rs[tid + st]; rq[tid] += rq[tid + st]; }
        __syncthreads();
    }
    const float mu   = rs[0] * (1.f / D_MODEL);
    const float var  = rq[0] * (1.f / D_MODEL) - mu * mu;
    const float rstd = rsqrtf(var + 1e-5f);
    for (int c = tid; c < D_MODEL; c += 256)
        hn[(size_t)m * D_MODEL + c] = (h16)((xr[c] - mu) * rstd * g[c] + b[c]);
}

// ---------------------------------------------------------------------------
// f32 -> f16 conversion (weights + intermediates)
// ---------------------------------------------------------------------------
__global__ __launch_bounds__(256)
void k_f2h(const float* __restrict__ s, h16* __restrict__ d, size_t n)
{
    size_t i = (size_t)blockIdx.x * 256 + threadIdx.x;
    if (i < n) d[i] = (h16)s[i];
}

// ---------------------------------------------------------------------------
// WMMA GEMM: C[M,N] = A[M,K] * W[N,K]^T  (f16 in, f32 accum)
// Block tile 256x64, BK=32, 256 threads = 8 waves. Wave w computes a 32x64
// strip: 2 A-fragments x 4 B-fragments -> 8 back-to-back WMMAs per K-step.
// Double-buffered LDS staging via global_load_async_to_lds_b128: tile t+1 is
// issued before waiting on tile t (s_wait_asynccnt 0x5 retires only the 5
// oldest per-wave async ops), overlapping HBM latency with the WMMA burst.
// Requires M % 256 == 0 and K % 32 == 0 (true for all call sites here).
// ---------------------------------------------------------------------------
template<int HALF_OUT>
__global__ __launch_bounds__(256)
void k_gemm(const h16* __restrict__ A, int lda,
            const h16* __restrict__ W, int ldw,
            float* __restrict__ Cf, h16* __restrict__ Ch, int ldc,
            int M, int N, int K)
{
    __shared__ h16 As[2][256 * 32];   // 2 x 16 KB
    __shared__ h16 Bs[2][64 * 32];    // 2 x  4 KB
    const int tid  = threadIdx.x;
    const int lane = tid & 31;
    const int wv   = tid >> 5;
    const int m0   = blockIdx.y * 256;
    const int n0   = blockIdx.x * 64;

    const uint32_t asb[2] = { (uint32_t)(uintptr_t)&As[0][0],
                              (uint32_t)(uintptr_t)&As[1][0] };
    const uint32_t bsb[2] = { (uint32_t)(uintptr_t)&Bs[0][0],
                              (uint32_t)(uintptr_t)&Bs[1][0] };

    // Issue one 256x32 A tile + 64x32 W tile: exactly 5 async ops per wave
    // (W issue is unconditional with a clamped row so ASYNCcnt stays uniform).
    auto issue_tile = [&](int k0, int buf) {
#pragma unroll
        for (int s = 0; s < 4; ++s) {
            int slot = tid + s * 256;
            int r = slot >> 2, c = (slot & 3) << 3;
            async_b128(asb[buf] + (uint32_t)((r * 32 + c) * 2),
                       A + (size_t)(m0 + r) * lda + (k0 + c));
        }
        int r = tid >> 2, c = (tid & 3) << 3;
        int gn = n0 + r; if (gn >= N) gn = N - 1;   // clamp, not skip
        async_b128(bsb[buf] + (uint32_t)((r * 32 + c) * 2),
                   W + (size_t)gn * ldw + (k0 + c));
    };

    const v8f vzero = {0.f, 0.f, 0.f, 0.f, 0.f, 0.f, 0.f, 0.f};
    v8f acc[2][4];
#pragma unroll
    for (int i = 0; i < 2; ++i)
#pragma unroll
        for (int j = 0; j < 4; ++j) acc[i][j] = vzero;

    const int row = lane & 15;          // fragment row / col within 16
    const int kb  = (lane >> 4) * 16;   // half-wave K split
    const int T   = K >> 5;             // number of 32-wide K tiles

    issue_tile(0, 0);                   // prologue: tile 0 -> buffer 0
    int cur = 0;
    for (int t = 0; t < T; ++t) {
        __syncthreads();                // all waves done reading buf[cur^1]
        if (t + 1 < T) {
            issue_tile((t + 1) << 5, cur ^ 1);
            wait_async5();              // tile t landed; tile t+1 in flight
        } else {
            wait_async0();
        }
        __syncthreads();                // tile t visible to all waves

        // hoist all fragments from buf[cur], then burst 8 WMMAs
        v16h af[2], bf[4];
#pragma unroll
        for (int i = 0; i < 2; ++i)
            af[i] = *(const v16h*)(&As[cur][(wv * 32 + i * 16 + row) * 32 + kb]);
#pragma unroll
        for (int j = 0; j < 4; ++j)
            bf[j] = *(const v16h*)(&Bs[cur][(j * 16 + row) * 32 + kb]);
#pragma unroll
        for (int j = 0; j < 4; ++j) {
            acc[0][j] = WMMA_F16(af[0], bf[j], acc[0][j]);
            acc[1][j] = WMMA_F16(af[1], bf[j], acc[1][j]);
        }
        cur ^= 1;
    }

    // epilogue: VGPR v holds M = v + 8*(lane>=16), N = lane&15
    const int nlane = lane & 15;
#pragma unroll
    for (int i = 0; i < 2; ++i) {
        const int mbase = m0 + wv * 32 + i * 16 + (lane >> 4) * 8;
#pragma unroll
        for (int j = 0; j < 4; ++j) {
            int gn = n0 + j * 16 + nlane;
            if (gn >= N) continue;
#pragma unroll
            for (int v = 0; v < 8; ++v) {
                size_t gm = (size_t)(mbase + v);
                if (HALF_OUT) Ch[gm * ldc + gn] = (h16)acc[i][j][v];
                else          Cf[gm * ldc + gn] = acc[i][j][v];
            }
        }
    }
}

// ---------------------------------------------------------------------------
// Depthwise causal conv (D_CONV=4) + SiLU. rev=1 runs anti-causally, which is
// exactly flip->conv->flip of the reference's bwd branch.
// ---------------------------------------------------------------------------
__global__ __launch_bounds__(256)
void k_conv_silu(const h16* __restrict__ xz, const float* __restrict__ cw,
                 const float* __restrict__ cb, h16* __restrict__ u, int rev)
{
    size_t idx = (size_t)blockIdx.x * 256 + threadIdx.x;
    if (idx >= (size_t)MTOK * D_INNER) return;
    int d = (int)(idx % D_INNER);
    int l = (int)((idx / D_INNER) % SEQ);
    int b = (int)(idx / ((size_t)D_INNER * SEQ));
    float acc = cb[d];
#pragma unroll
    for (int j = 0; j < D_CONV; ++j) {
        int lp = rev ? (l + (D_CONV - 1) - j) : (l - (D_CONV - 1) + j);
        if (lp >= 0 && lp < SEQ)
            acc += cw[d * D_CONV + j] *
                   (float)xz[((size_t)b * SEQ + lp) * (2 * D_INNER) + d];
    }
    u[idx] = (h16)silu_f(acc);
}

// ---------------------------------------------------------------------------
// dt = softplus(dt_pre + dt_b)
// ---------------------------------------------------------------------------
__global__ __launch_bounds__(256)
void k_softplus(float* __restrict__ dt, const float* __restrict__ dtb)
{
    size_t idx = (size_t)blockIdx.x * 256 + threadIdx.x;
    if (idx >= (size_t)MTOK * D_INNER) return;
    int d = (int)(idx % D_INNER);
    float v = dt[idx] + dtb[d];
    dt[idx] = (v > 20.f) ? v : log1pf(__expf(v));
}

// ---------------------------------------------------------------------------
// Selective scan: thread = (batch, channel), 16 states in registers.
// B_t/C_t (shared across channels) staged in 32-step LDS chunks.
// rev=1 walks the sequence backwards (flip->scan->flip of bwd branch).
// ---------------------------------------------------------------------------
__global__ __launch_bounds__(256)
void k_scan(const float* __restrict__ dt, const h16* __restrict__ u,
            const float* __restrict__ dbc, const float* __restrict__ A_log,
            float* __restrict__ ys, int rev)
{
    const int d = blockIdx.x * 256 + threadIdx.x;   // 0..D_INNER-1
    const int b = blockIdx.y;
    float a[D_STATE], h[D_STATE];
#pragma unroll
    for (int n = 0; n < D_STATE; ++n) {
        a[n] = -__expf(A_log[(size_t)d * D_STATE + n]);
        h[n] = 0.f;
    }
    __shared__ float sBC[32 * 32];   // 32 steps x (16 B + 16 C)
    for (int c0 = 0; c0 < SEQ; c0 += 32) {
        __syncthreads();
        for (int s = threadIdx.x; s < 32 * 32; s += 256) {
            int step = s >> 5, i = s & 31;
            int l = rev ? (SEQ - 1 - (c0 + step)) : (c0 + step);
            sBC[s] = dbc[((size_t)b * SEQ + l) * (DT_RANK + 2 * D_STATE) + DT_RANK + i];
        }
        __syncthreads();
        for (int st = 0; st < 32; ++st) {
            int l = rev ? (SEQ - 1 - (c0 + st)) : (c0 + st);
            size_t m = (size_t)b * SEQ + l;
            float dt_t = dt[m * D_INNER + d];
            float u_t  = (float)u[m * D_INNER + d];
            float du   = dt_t * u_t;
            const float* bc = &sBC[st * 32];
            float y = 0.f;
#pragma unroll
            for (int n = 0; n < D_STATE; ++n) {
                h[n] = __expf(dt_t * a[n]) * h[n] + du * bc[n];
                y += h[n] * bc[D_STATE + n];
            }
            ys[m * D_INNER + d] = y;
        }
    }
}

// ---------------------------------------------------------------------------
// Gating: yg = (ys + u*D) * silu(z), written f16 for the out-proj GEMM
// ---------------------------------------------------------------------------
__global__ __launch_bounds__(256)
void k_gate(const float* __restrict__ ys, const h16* __restrict__ u,
            const float* __restrict__ Dp, const h16* __restrict__ xz,
            h16* __restrict__ yg)
{
    size_t idx = (size_t)blockIdx.x * 256 + threadIdx.x;
    if (idx >= (size_t)MTOK * D_INNER) return;
    int d = (int)(idx % D_INNER);
    size_t m = idx / D_INNER;
    float z = (float)xz[m * (2 * D_INNER) + D_INNER + d];
    float y = ys[idx] + (float)u[idx] * Dp[d];
    yg[idx] = (h16)(y * silu_f(z));
}

// ---------------------------------------------------------------------------
// Residual + bias: out = x + acc + proj_b
// ---------------------------------------------------------------------------
__global__ __launch_bounds__(256)
void k_residual(const float* __restrict__ x, const float* __restrict__ acc,
                const float* __restrict__ pb, float* __restrict__ out)
{
    size_t idx = (size_t)blockIdx.x * 256 + threadIdx.x;
    if (idx >= (size_t)MTOK * D_MODEL) return;
    int c = (int)(idx % D_MODEL);
    out[idx] = x[idx] + acc[idx] + pb[c];
}

// ---------------------------------------------------------------------------
extern "C" void kernel_launch(void* const* d_in, const int* in_sizes, int n_in,
                              void* d_out, int out_size, void* d_ws, size_t ws_size,
                              hipStream_t stream)
{
    (void)in_sizes; (void)n_in; (void)out_size; (void)ws_size;
    const float* x      = (const float*)d_in[0];
    const float* ln_g   = (const float*)d_in[1];
    const float* ln_b   = (const float*)d_in[2];
    const float* proj_w = (const float*)d_in[21];
    const float* proj_b = (const float*)d_in[22];
    float* out = (float*)d_out;

    char* ws = (char*)d_ws;
    size_t off = 0;
    auto alloc = [&](size_t bytes) -> char* {
        char* p = ws + off;
        off += (bytes + 255) & ~(size_t)255;
        return p;
    };
    h16*   hnh  = (h16*)  alloc((size_t)MTOK * D_MODEL * 2);
    h16*   wh   = (h16*)  alloc((size_t)2 * D_INNER * D_MODEL * 2);   // weight staging
    h16*   xzh  = (h16*)  alloc((size_t)MTOK * 2 * D_INNER * 2);
    h16*   uh   = (h16*)  alloc((size_t)MTOK * D_INNER * 2);
    float* dbcf = (float*)alloc((size_t)MTOK * (DT_RANK + 2 * D_STATE) * 4);
    h16*   dbch = (h16*)  alloc((size_t)MTOK * (DT_RANK + 2 * D_STATE) * 2);
    float* dtf  = (float*)alloc((size_t)MTOK * D_INNER * 4);
    float* ysf  = (float*)alloc((size_t)MTOK * D_INNER * 4);
    h16*   ygh  = (h16*)  alloc((size_t)MTOK * D_INNER * 2);
    h16*   ycat = (h16*)  alloc((size_t)MTOK * 2 * D_MODEL * 2);
    float* accf = dtf;    // dtf dead after branch 2's scan; reuse for final proj

    const dim3 blk(256);
    const size_t ED = (size_t)MTOK * D_INNER;
    const unsigned gED = (unsigned)((ED + 255) / 256);

    auto cvt = [&](const float* s, h16* d, size_t n) {
        k_f2h<<<dim3((unsigned)((n + 255) / 256)), blk, 0, stream>>>(s, d, n);
    };

    k_layernorm<<<dim3(MTOK), blk, 0, stream>>>(x, ln_g, ln_b, hnh);

    for (int br = 0; br < 2; ++br) {
        const int base = 3 + br * 9;
        const float* in_w  = (const float*)d_in[base + 0];
        const float* cw    = (const float*)d_in[base + 1];
        const float* cb    = (const float*)d_in[base + 2];
        const float* xp_w  = (const float*)d_in[base + 3];
        const float* dt_w  = (const float*)d_in[base + 4];
        const float* dt_b  = (const float*)d_in[base + 5];
        const float* A_log = (const float*)d_in[base + 6];
        const float* Dp    = (const float*)d_in[base + 7];
        const float* out_w = (const float*)d_in[base + 8];

        // xz = hn @ in_w^T   (4096 x 4096 x 1024)
        cvt(in_w, wh, (size_t)2 * D_INNER * D_MODEL);
        k_gemm<1><<<dim3((2 * D_INNER) / 64, MTOK / 256), blk, 0, stream>>>(
            hnh, D_MODEL, wh, D_MODEL, nullptr, xzh, 2 * D_INNER,
            MTOK, 2 * D_INNER, D_MODEL);

        // depthwise conv + silu (anti-causal for bwd branch)
        k_conv_silu<<<dim3(gED), blk, 0, stream>>>(xzh, cw, cb, uh, br);

        // dbc = u @ xproj^T   (4096 x 96 x 2048)
        cvt(xp_w, wh, (size_t)(DT_RANK + 2 * D_STATE) * D_INNER);
        k_gemm<0><<<dim3((96 + 63) / 64, MTOK / 256), blk, 0, stream>>>(
            uh, D_INNER, wh, D_INNER, dbcf, nullptr, DT_RANK + 2 * D_STATE,
            MTOK, DT_RANK + 2 * D_STATE, D_INNER);
        cvt(dbcf, dbch, (size_t)MTOK * (DT_RANK + 2 * D_STATE));

        // dt_pre = dbc[:, :64] @ dt_w^T   (4096 x 2048 x 64)
        cvt(dt_w, wh, (size_t)D_INNER * DT_RANK);
        k_gemm<0><<<dim3(D_INNER / 64, MTOK / 256), blk, 0, stream>>>(
            dbch, DT_RANK + 2 * D_STATE, wh, DT_RANK, dtf, nullptr, D_INNER,
            MTOK, D_INNER, DT_RANK);
        k_softplus<<<dim3(gED), blk, 0, stream>>>(dtf, dt_b);

        // selective scan (backwards for bwd branch)
        k_scan<<<dim3(D_INNER / 256, BATCH), blk, 0, stream>>>(
            dtf, uh, dbcf, A_log, ysf, br);

        // gate + out-proj into concat slot (flip already folded into rev scan)
        k_gate<<<dim3(gED), blk, 0, stream>>>(ysf, uh, Dp, xzh, ygh);
        cvt(out_w, wh, (size_t)D_MODEL * D_INNER);
        k_gemm<1><<<dim3(D_MODEL / 64, MTOK / 256), blk, 0, stream>>>(
            ygh, D_INNER, wh, D_INNER, nullptr, ycat + br * D_MODEL, 2 * D_MODEL,
            MTOK, D_MODEL, D_INNER);
    }

    // y = concat @ proj_w^T ; out = x + y + proj_b
    cvt(proj_w, wh, (size_t)D_MODEL * 2 * D_MODEL);
    k_gemm<0><<<dim3(D_MODEL / 64, MTOK / 256), blk, 0, stream>>>(
        ycat, 2 * D_MODEL, wh, 2 * D_MODEL, accf, nullptr, D_MODEL,
        MTOK, D_MODEL, 2 * D_MODEL);
    k_residual<<<dim3((unsigned)(((size_t)MTOK * D_MODEL + 255) / 256)), blk, 0, stream>>>(
        x, accf, proj_b, out);
}
